// Adaptive_Score_Generation_55525337203208
// MI455X (gfx1250) — compile-verified, matrix operand order checked by reading, not run on hardware
//
#include <hip/hip_runtime.h>
#include <hip/hip_bf16.h>
#include <math.h>

// Problem constants (match reference)
#define DIMC     512
#define D_STATE  16
#define D_CONV   4
#define NSLICES  5
#define D_INNER  1024
#define E2       2048          // 2*D_INNER
#define DT_RANK  32
#define KPROJ    64            // DT_RANK + 2*D_STATE
#define B_SZ     2
#define SEQ      5120
#define TTOK     (B_SZ*SEQ)    // 10240 tokens
#define LSL      (SEQ/NSLICES) // 1024
#define LGS      2048

typedef __attribute__((ext_vector_type(16))) __bf16 v16bf;
typedef __attribute__((ext_vector_type(8)))  __bf16 v8bf;
typedef __attribute__((ext_vector_type(8)))  float  v8f;

__device__ __forceinline__ __bf16 f2bf(float f) {
  unsigned u = __builtin_bit_cast(unsigned, f);
  unsigned r = u + 0x7FFFu + ((u >> 16) & 1u);   // round-to-nearest-even
  unsigned short s = (unsigned short)(r >> 16);
  __bf16 h;
  __builtin_memcpy(&h, &s, sizeof(h));
  return h;
}

__device__ __forceinline__ float sigmoidf_(float x) { return 1.0f / (1.0f + __expf(-x)); }
__device__ __forceinline__ float siluf_(float x)    { return x * sigmoidf_(x); }
__device__ __forceinline__ float softplusf_(float x){ return (x > 20.0f) ? x : log1pf(__expf(x)); }

// Branch index maps: virtual position p (0..SEQ-1) -> original l.
__device__ __forceinline__ int rmap_l(int br, int p) {
  if (br == 0) return p;
  if (br == 1) return SEQ - 1 - p;
  return (p % NSLICES) * LSL + (p / NSLICES);
}
__device__ __forceinline__ int wmap_l(int br, int p) {
  if (br == 0) return p;
  if (br == 1) return SEQ - 1 - p;
  return (p % LSL) * NSLICES + (p / LSL);
}

// ---------------- f32 -> bf16 bulk convert --------------------------------
__global__ void cvt_bf16_kernel(const float* __restrict__ src,
                                __bf16* __restrict__ dst, int n) {
  int i = blockIdx.x * 256 + threadIdx.x;
  if (i < n) dst[i] = f2bf(src[i]);
}

// ---------------- LayerNorm + ReLU: x[T,512] -> hbf[T,512] (bf16) ---------
__global__ void ln_relu_kernel(const float* __restrict__ x,
                               const float* __restrict__ g,
                               const float* __restrict__ b,
                               __bf16* __restrict__ hbf) {
  int t    = blockIdx.x * 8 + (threadIdx.x >> 5);   // one wave32 per token
  int lane = threadIdx.x & 31;
  const float* xr = x + (size_t)t * DIMC;
  float sum = 0.f, sq = 0.f;
  for (int c = lane; c < DIMC; c += 32) { float v = xr[c]; sum += v; sq += v * v; }
  for (int o = 16; o > 0; o >>= 1) {
    sum += __shfl_xor(sum, o, 32);
    sq  += __shfl_xor(sq,  o, 32);
  }
  float mu = sum * (1.0f / DIMC);
  float var = sq * (1.0f / DIMC) - mu * mu;
  float rs = rsqrtf(var + 1e-5f);
  __bf16* hr = hbf + (size_t)t * DIMC;
  for (int c = lane; c < DIMC; c += 32) {
    float v = (xr[c] - mu) * rs * g[c] + b[c];
    hr[c] = f2bf(v > 0.f ? v : 0.f);
  }
}

// ---------------- WMMA GEMM: C[M,N] = A[M,K] * W[N,K]^T (bf16 in, f32 out)
// A,W pre-converted bf16. Per lane, each 16x32 fragment is two b128 loads
// (K = hs*8..+7 and 16+hs*8..+7 are contiguous in the CDNA5 VGPR layout).
// Each wave computes a 16x32 output (one A fragment, two B fragments ->
// two v_wmma per K step). Optionally writes a bf16 copy of C (Cbf).
// Grid: (Mtiles * Ntiles/2) / 8 blocks of 256 threads (exact).
__global__ void wmma_gemm_kernel(const __bf16* __restrict__ A, int lda,
                                 const __bf16* __restrict__ W, int ldb,
                                 float* __restrict__ C, int ldc,
                                 __bf16* __restrict__ Cbf,
                                 int K, int NtilesHalf) {
  int wave = blockIdx.x * 8 + (threadIdx.x >> 5);
  int mt  = wave / NtilesHalf;
  int nt2 = (wave % NtilesHalf) * 2;
  int lane = threadIdx.x & 31;
  int mn = lane & 15;        // A row / C col / W row
  int hs = lane >> 4;        // K half-select
  const __bf16* Arow = A + (size_t)(mt * 16 + mn) * lda;
  const __bf16* W0   = W + (size_t)(nt2 * 16 + mn) * ldb;
  const __bf16* W1   = W0 + (size_t)16 * ldb;
  v8f c0 = {}, c1 = {};
  for (int kk = 0; kk < K; kk += 32) {
    int o0 = kk + hs * 8;
    int o1 = kk + 16 + hs * 8;
    __builtin_prefetch(Arow + kk + 64, 0, 1);   // global_prefetch_b8 next tile
    __builtin_prefetch(W0 + kk + 64, 0, 1);
    __builtin_prefetch(W1 + kk + 64, 0, 1);
    v8bf alo = *(const v8bf*)(Arow + o0);
    v8bf ahi = *(const v8bf*)(Arow + o1);
    v8bf b0l = *(const v8bf*)(W0 + o0);
    v8bf b0h = *(const v8bf*)(W0 + o1);
    v8bf b1l = *(const v8bf*)(W1 + o0);
    v8bf b1h = *(const v8bf*)(W1 + o1);
    v16bf a  = __builtin_shufflevector(alo, ahi, 0,1,2,3,4,5,6,7,8,9,10,11,12,13,14,15);
    v16bf b0 = __builtin_shufflevector(b0l, b0h, 0,1,2,3,4,5,6,7,8,9,10,11,12,13,14,15);
    v16bf b1 = __builtin_shufflevector(b1l, b1h, 0,1,2,3,4,5,6,7,8,9,10,11,12,13,14,15);
    c0 = __builtin_amdgcn_wmma_f32_16x16x32_bf16(false, a, false, b0,
                                                 (short)0, c0, false, false);
    c1 = __builtin_amdgcn_wmma_f32_16x16x32_bf16(false, a, false, b1,
                                                 (short)0, c1, false, false);
  }
#pragma unroll
  for (int j = 0; j < 8; ++j) {   // C/D layout: VGPR j -> M = j + 8*hs, N = lane&15
    size_t m = (size_t)(mt * 16 + j + 8 * hs);
    C[m * ldc + nt2 * 16 + mn]        = c0[j];
    C[m * ldc + (nt2 + 1) * 16 + mn]  = c1[j];
  }
  if (Cbf) {                       // kernel-arg uniform: EXEC stays full
#pragma unroll
    for (int j = 0; j < 8; ++j) {
      size_t m = (size_t)(mt * 16 + j + 8 * hs);
      Cbf[m * ldc + nt2 * 16 + mn]       = f2bf(c0[j]);
      Cbf[m * ldc + (nt2 + 1) * 16 + mn] = f2bf(c1[j]);
    }
  }
}

// ---------------- Depthwise conv(4) + SiLU over virtual sequence ----------
__global__ void conv_silu_kernel(const float* __restrict__ xz,
                                 const float* __restrict__ w,
                                 const float* __restrict__ bias,
                                 float* __restrict__ xcv,
                                 __bf16* __restrict__ xcvbf, int br) {
  size_t idx = (size_t)blockIdx.x * 256 + threadIdx.x;  // over TTOK*D_INNER
  int d = (int)(idx & (D_INNER - 1));
  size_t vr = idx >> 10;             // virtual token row
  int b = (int)(vr / SEQ);
  int p = (int)(vr % SEQ);
  float s = bias[d];
#pragma unroll
  for (int j = 0; j < D_CONV; ++j) {
    int q = p - (D_CONV - 1) + j;
    if (q >= 0) {
      size_t t = (size_t)b * SEQ + rmap_l(br, q);
      s += w[d * D_CONV + j] * xz[t * E2 + d];
    }
  }
  float v = siluf_(s);
  xcv[vr * D_INNER + d] = v;
  xcvbf[vr * D_INNER + d] = f2bf(v);
}

// ---------------- Selective scan + D skip + SiLU(z) gate ------------------
__global__ void scan_kernel(const float* __restrict__ xz,
                            const float* __restrict__ xcv,
                            const float* __restrict__ dtp,
                            const float* __restrict__ xdb,
                            const float* __restrict__ dt_bias,
                            const float* __restrict__ A_log,
                            const float* __restrict__ Dv,
                            float* __restrict__ acc, int br) {
  int b = blockIdx.x >> 2;
  int d = ((blockIdx.x & 3) << 8) + threadIdx.x;
  __shared__ float Bs[D_STATE];
  __shared__ float Cs[D_STATE];
  float Ac[D_STATE], h[D_STATE];
#pragma unroll
  for (int n = 0; n < D_STATE; ++n) {
    Ac[n] = -__expf(A_log[d * D_STATE + n]);
    h[n] = 0.f;
  }
  float dbias = dt_bias[d];
  float Dd = Dv[d];
  for (int p = 0; p < SEQ; ++p) {
    size_t row = (size_t)b * SEQ + p;   // virtual row
    __syncthreads();
    if (threadIdx.x < 2 * D_STATE) {
      float v = xdb[row * KPROJ + DT_RANK + threadIdx.x];
      if (threadIdx.x < D_STATE) Bs[threadIdx.x] = v;
      else                       Cs[threadIdx.x - D_STATE] = v;
    }
    __syncthreads();
    float dt = softplusf_(dtp[row * D_INNER + d] + dbias);
    float u = xcv[row * D_INNER + d];
    float y = 0.f;
#pragma unroll
    for (int n = 0; n < D_STATE; ++n) {
      h[n] = __expf(dt * Ac[n]) * h[n] + dt * Bs[n] * u;
      y += h[n] * Cs[n];
    }
    y += u * Dd;
    size_t tz = (size_t)b * SEQ + rmap_l(br, p);
    float z = xz[tz * E2 + D_INNER + d];
    size_t tw = (size_t)b * SEQ + wmap_l(br, p);
    acc[tw * D_INNER + d] += y * siluf_(z);
  }
}

// ---------------- Final linear + sigmoid per token ------------------------
__global__ void score_kernel(const float* __restrict__ yo,
                             const float* __restrict__ lw,
                             const float* __restrict__ lb,
                             float* __restrict__ out_ind) {
  int t = blockIdx.x * 8 + (threadIdx.x >> 5);
  int lane = threadIdx.x & 31;
  const float* yr = yo + (size_t)t * DIMC;
  float s = 0.f;
  for (int c = lane; c < DIMC; c += 32) s += yr[c] * lw[c];
  for (int o = 16; o > 0; o >>= 1) s += __shfl_xor(s, o, 32);
  if (lane == 0) out_ind[t] = sigmoidf_(s + lb[0]);
}

// ---------------- Group scores: linear interp (align_corners) + sigmoid ---
__global__ void group_kernel(const float* __restrict__ g0,
                             const float* __restrict__ g1,
                             const float* __restrict__ g2,
                             float* __restrict__ out) {
  int i = blockIdx.x * 256 + threadIdx.x;
  if (i >= 3 * SEQ) return;
  int g = i / SEQ;
  int j = i % SEQ;
  const float* v = (g == 0) ? g0 : (g == 1) ? g1 : g2;
  float pos = (float)j * ((float)(LGS - 1) / (float)(SEQ - 1));
  int i0 = (int)floorf(pos);
  if (i0 < 0) i0 = 0;
  if (i0 > LGS - 2) i0 = LGS - 2;
  float w = pos - (float)i0;
  out[i] = sigmoidf_(v[i0] * (1.f - w) + v[i0 + 1] * w);
}

extern "C" void kernel_launch(void* const* d_in, const int* in_sizes, int n_in,
                              void* d_out, int out_size, void* d_ws, size_t ws_size,
                              hipStream_t stream) {
  const float* x     = (const float*)d_in[0];
  const float* lng   = (const float*)d_in[1];
  const float* lnb   = (const float*)d_in[2];
  const float* w_in  = (const float*)d_in[3];   // [2048, 512]
  const float* w_out = (const float*)d_in[4];   // [512, 1024]
  const float* lin_w = (const float*)d_in[5];   // [1, 512]
  const float* lin_b = (const float*)d_in[6];
  const float* g0 = (const float*)d_in[7];
  const float* g1 = (const float*)d_in[8];
  const float* g2 = (const float*)d_in[9];

  // Workspace layout (~292 MB total)
  char* ws = (char*)d_ws;
  size_t off = 0;
  auto alloc = [&](size_t bytes) {
    void* p = ws + off;
    off += (bytes + 255) & ~(size_t)255;
    return p;
  };
  float*  xz    = (float*)alloc((size_t)TTOK * E2 * 4);
  float*  xcv   = (float*)alloc((size_t)TTOK * D_INNER * 4);
  float*  dtp   = (float*)alloc((size_t)TTOK * D_INNER * 4);
  float*  acc   = (float*)alloc((size_t)TTOK * D_INNER * 4);
  float*  hb    = (float*)alloc((size_t)TTOK * DIMC * 4);      // y_out f32
  float*  xdb   = (float*)alloc((size_t)TTOK * KPROJ * 4);
  __bf16* hbf   = (__bf16*)alloc((size_t)TTOK * DIMC * 2);
  __bf16* xcvbf = (__bf16*)alloc((size_t)TTOK * D_INNER * 2);
  __bf16* accbf = (__bf16*)alloc((size_t)TTOK * D_INNER * 2);
  __bf16* xdbbf = (__bf16*)alloc((size_t)TTOK * KPROJ * 2);
  __bf16* winbf = (__bf16*)alloc((size_t)E2 * DIMC * 2);
  __bf16* woutbf= (__bf16*)alloc((size_t)DIMC * D_INNER * 2);
  __bf16* xpjbf = (__bf16*)alloc((size_t)KPROJ * D_INNER * 2);
  __bf16* dtwbf = (__bf16*)alloc((size_t)D_INNER * DT_RANK * 2);

  hipMemsetAsync(acc, 0, (size_t)TTOK * D_INNER * 4, stream);

  // Weight converts (per call; no caching allowed)
  cvt_bf16_kernel<<<(E2 * DIMC + 255) / 256, 256, 0, stream>>>(w_in, winbf, E2 * DIMC);
  cvt_bf16_kernel<<<(DIMC * D_INNER + 255) / 256, 256, 0, stream>>>(w_out, woutbf,
                                                                    DIMC * D_INNER);

  ln_relu_kernel<<<TTOK / 8, 256, 0, stream>>>(x, lng, lnb, hbf);

  // in_proj: [10240,512] x [2048,512]^T -> xz [10240,2048]
  {
    int mt = TTOK / 16, nth = (E2 / 16) / 2;                  // 640 x 64
    wmma_gemm_kernel<<<mt * nth / 8, 256, 0, stream>>>(hbf, DIMC, winbf, DIMC,
                                                       xz, E2, nullptr, DIMC, nth);
  }

  for (int br = 0; br < 3; ++br) {
    const float* conv_w  = (const float*)d_in[10 + br * 7 + 0];
    const float* conv_b  = (const float*)d_in[10 + br * 7 + 1];
    const float* xproj_w = (const float*)d_in[10 + br * 7 + 2];
    const float* dt_w    = (const float*)d_in[10 + br * 7 + 3];
    const float* dt_bias = (const float*)d_in[10 + br * 7 + 4];
    const float* A_log   = (const float*)d_in[10 + br * 7 + 5];
    const float* Dv      = (const float*)d_in[10 + br * 7 + 6];

    cvt_bf16_kernel<<<(KPROJ * D_INNER + 255) / 256, 256, 0, stream>>>(
        xproj_w, xpjbf, KPROJ * D_INNER);
    cvt_bf16_kernel<<<(D_INNER * DT_RANK + 255) / 256, 256, 0, stream>>>(
        dt_w, dtwbf, D_INNER * DT_RANK);

    conv_silu_kernel<<<(TTOK * D_INNER) / 256, 256, 0, stream>>>(xz, conv_w, conv_b,
                                                                 xcv, xcvbf, br);
    // x_proj: [10240,1024] x [64,1024]^T -> xdb [10240,64] (+ bf16 copy)
    {
      int mt = TTOK / 16, nth = (KPROJ / 16) / 2;             // 640 x 2
      wmma_gemm_kernel<<<mt * nth / 8, 256, 0, stream>>>(xcvbf, D_INNER, xpjbf,
                                                         D_INNER, xdb, KPROJ,
                                                         xdbbf, D_INNER, nth);
    }
    // dt proj: [10240,32] (cols 0..31 of xdbbf) x [1024,32]^T -> dtp [10240,1024]
    {
      int mt = TTOK / 16, nth = (D_INNER / 16) / 2;           // 640 x 32
      wmma_gemm_kernel<<<mt * nth / 8, 256, 0, stream>>>(xdbbf, KPROJ, dtwbf,
                                                         DT_RANK, dtp, D_INNER,
                                                         nullptr, DT_RANK, nth);
    }
    scan_kernel<<<B_SZ * 4, 256, 0, stream>>>(xz, xcv, dtp, xdb, dt_bias,
                                              A_log, Dv, acc, br);
  }

  // out_proj: [10240,1024] x [512,1024]^T -> hb [10240,512]
  cvt_bf16_kernel<<<(TTOK * D_INNER) / 256, 256, 0, stream>>>(acc, accbf,
                                                              TTOK * D_INNER);
  {
    int mt = TTOK / 16, nth = (DIMC / 16) / 2;                // 640 x 16
    wmma_gemm_kernel<<<mt * nth / 8, 256, 0, stream>>>(accbf, D_INNER, woutbf,
                                                       D_INNER, hb, DIMC,
                                                       nullptr, D_INNER, nth);
  }

  float* outp = (float*)d_out;   // [0,15360) group scores, [15360,25600) individual
  score_kernel<<<TTOK / 8, 256, 0, stream>>>(hb, lin_w, lin_b, outp + 3 * SEQ);
  group_kernel<<<(3 * SEQ + 255) / 256, 256, 0, stream>>>(g0, g1, g2, outp);
}
